// MambaEnc_78683800862963
// MI455X (gfx1250) — compile-verified, matrix operand order checked by reading, not run on hardware
//
#include <hip/hip_runtime.h>

// ---------------------------------------------------------------------------
// MI455X (gfx1250, wave32) implementation.
// All GEMM-shaped work (convs via im2col, all Mamba projections) runs on
// v_wmma_f32_16x16x32_bf16 with f32 accumulation. Each wave computes a 2x2
// grid of 16x16 tiles (32x32 output) so every A/B fragment feeds two WMMAs.
// Fragment loads follow the CDNA5 ISA 16-bit A/B operand layouts (two
// contiguous 16B K-chunks per lane).
// ---------------------------------------------------------------------------

typedef unsigned short u16;
typedef unsigned int   u32;
typedef __attribute__((ext_vector_type(16))) __bf16 v16bf;
typedef __attribute__((ext_vector_type(8)))  float  v8f;
typedef __attribute__((ext_vector_type(4)))  u32    u32x4;

#define BN_S 0.9999950000374997f   // 1/sqrt(1+1e-5)

__device__ __forceinline__ u16 f2bf(float f) {
  union { float f; u32 u; } v; v.f = f;
  u32 r = v.u + 0x7FFFu + ((v.u >> 16) & 1u);
  return (u16)(r >> 16);
}
__device__ __forceinline__ float siluf(float x) { return x / (1.f + __expf(-x)); }
__device__ __forceinline__ float leakyf(float x) { return x >= 0.f ? x : 0.2f * x; }
__device__ __forceinline__ float epilogue(float v, int act) {
  if (act == 1) return leakyf(v);
  if (act == 2) return (v > 20.f) ? v : log1pf(__expf(v));
  return v;
}

// ---------------------------------------------------------------------------
// WMMA GEMM: C[M,Npad] = act(A[M,Kpad] * W[Npad,Kpad]^T + bias)
// A, W bf16 row-major; one 32x32 register-blocked tile per wave (4 WMMAs
// per K-step, 2x operand reuse); 8 waves per block. M % 32 == 0, Npad % 32 == 0.
// ---------------------------------------------------------------------------
__global__ __launch_bounds__(256) void k_gemm_bf16(
    const u16* __restrict__ A, int lda,
    const u16* __restrict__ W, int ldw,
    float* __restrict__ C, int ldc,
    const float* __restrict__ bias, int nbias,
    int M, int Npad, int Kpad, int act)
{
  int wave = threadIdx.x >> 5, lane = threadIdx.x & 31;
  int tilesN = Npad >> 5;
  int t = blockIdx.x * 8 + wave;
  if (t >= (M >> 5) * tilesN) return;
  int tm = (t / tilesN) << 5, tn = (t % tilesN) << 5;
  int r = lane & 15, h = lane >> 4;
  const u16* a0 = A + (size_t)(tm + r) * lda + h * 8;
  const u16* a1 = a0 + (size_t)16 * lda;
  const u16* b0 = W + (size_t)(tn + r) * ldw + h * 8;
  const u16* b1 = b0 + (size_t)16 * ldw;
  v8f c00 = {}, c01 = {}, c10 = {}, c11 = {};
  for (int k = 0; k < Kpad; k += 32) {
    union Fr { u32x4 q[2]; v16bf v; } fa0, fa1, fb0, fb1;
    fa0.q[0] = *(const u32x4*)(a0 + k); fa0.q[1] = *(const u32x4*)(a0 + k + 16);
    fa1.q[0] = *(const u32x4*)(a1 + k); fa1.q[1] = *(const u32x4*)(a1 + k + 16);
    fb0.q[0] = *(const u32x4*)(b0 + k); fb0.q[1] = *(const u32x4*)(b0 + k + 16);
    fb1.q[0] = *(const u32x4*)(b1 + k); fb1.q[1] = *(const u32x4*)(b1 + k + 16);
    __builtin_prefetch(a0 + k + 32, 0, 1);
    __builtin_prefetch(a1 + k + 32, 0, 1);
    __builtin_prefetch(b0 + k + 32, 0, 1);
    __builtin_prefetch(b1 + k + 32, 0, 1);
    c00 = __builtin_amdgcn_wmma_f32_16x16x32_bf16(false, fa0.v, false, fb0.v,
                                                  (short)0, c00, false, false);
    c01 = __builtin_amdgcn_wmma_f32_16x16x32_bf16(false, fa0.v, false, fb1.v,
                                                  (short)0, c01, false, false);
    c10 = __builtin_amdgcn_wmma_f32_16x16x32_bf16(false, fa1.v, false, fb0.v,
                                                  (short)0, c10, false, false);
    c11 = __builtin_amdgcn_wmma_f32_16x16x32_bf16(false, fa1.v, false, fb1.v,
                                                  (short)0, c11, false, false);
  }
  int n0 = tn + r, n1 = tn + 16 + r;
  float bv0 = (bias != nullptr && n0 < nbias) ? bias[n0] : 0.f;
  float bv1 = (bias != nullptr && n1 < nbias) ? bias[n1] : 0.f;
#pragma unroll
  for (int i = 0; i < 8; ++i) {
    size_t row0 = (size_t)(tm + h * 8 + i) * ldc;
    size_t row1 = (size_t)(tm + 16 + h * 8 + i) * ldc;
    C[row0 + n0] = epilogue(c00[i] + bv0, act);
    C[row0 + n1] = epilogue(c01[i] + bv1, act);
    C[row1 + n0] = epilogue(c10[i] + bv0, act);
    C[row1 + n1] = epilogue(c11[i] + bv1, act);
  }
}

// im2col from NHWC f32 -> bf16 [B*OH*OW, Kpad], optional fused BN (g,b).
__global__ void k_im2col(const float* __restrict__ in, u16* __restrict__ out,
                         const float* __restrict__ g, const float* __restrict__ bb,
                         int B, int H, int W, int C, int OH, int OW,
                         int kh, int kw, int stride, int pad, int dil, int Kpad,
                         long total)
{
  long idx = blockIdx.x * (long)blockDim.x + threadIdx.x;
  if (idx >= total) return;
  int kk = (int)(idx % Kpad);
  long rowi = idx / Kpad;
  int K = C * kh * kw;
  float v = 0.f;
  if (kk < K) {
    int ci = kk / (kh * kw);
    int rem = kk % (kh * kw);
    int ky = rem / kw, kx = rem % kw;
    int ow = (int)(rowi % OW); long t2 = rowi / OW;
    int oh = (int)(t2 % OH);  int b = (int)(t2 / OH);
    int iy = oh * stride - pad + ky * dil;
    int ix = ow * stride - pad + kx * dil;
    if (iy >= 0 && iy < H && ix >= 0 && ix < W) {
      v = in[(((long)b * H + iy) * W + ix) * C + ci];
      if (g != nullptr) v = v * (g[ci] * BN_S) + bb[ci];
    }
  }
  out[idx] = f2bf(v);
}

// weight f32 [N,K] -> bf16 [Npad,Kpad] zero-padded
__global__ void k_wcast(const float* __restrict__ w, u16* __restrict__ out,
                        int N, int K, int Npad, int Kpad, long total)
{
  long idx = blockIdx.x * (long)blockDim.x + threadIdx.x;
  if (idx >= total) return;
  int k = (int)(idx % Kpad); int n = (int)(idx / Kpad);
  float v = (n < N && k < K) ? w[(long)n * K + k] : 0.f;
  out[idx] = f2bf(v);
}

// activation f32 (strided column gather) -> bf16 [M,Kpad] zero-padded
__global__ void k_cast(const float* __restrict__ src, int ld, int off,
                       u16* __restrict__ out, int K, int Kpad, long total)
{
  long idx = blockIdx.x * (long)blockDim.x + threadIdx.x;
  if (idx >= total) return;
  int k = (int)(idx % Kpad); long m = idx / Kpad;
  float v = (k < K) ? src[m * (long)ld + off + k] : 0.f;
  out[idx] = f2bf(v);
}

// layernorm: one 256-thread block per row, D <= 256 (wave32 shfl + LDS reduce)
__global__ __launch_bounds__(256) void k_layernorm(
    const float* __restrict__ x, int ldx, float* __restrict__ o, int ldo,
    const float* __restrict__ w, const float* __restrict__ b, int D)
{
  __shared__ float red[8];
  long row = blockIdx.x;
  int t = threadIdx.x;
  float v = (t < D) ? x[row * (long)ldx + t] : 0.f;
  float s = v;
#pragma unroll
  for (int m = 16; m; m >>= 1) s += __shfl_xor(s, m, 32);
  if ((t & 31) == 31) red[t >> 5] = s;
  __syncthreads();
  float tot = 0.f;
#pragma unroll
  for (int i = 0; i < 8; ++i) tot += red[i];
  float mean = tot / D;
  __syncthreads();
  float d = (t < D) ? (v - mean) : 0.f;
  float s2 = d * d;
#pragma unroll
  for (int m = 16; m; m >>= 1) s2 += __shfl_xor(s2, m, 32);
  if ((t & 31) == 31) red[t >> 5] = s2;
  __syncthreads();
  float tot2 = 0.f;
#pragma unroll
  for (int i = 0; i < 8; ++i) tot2 += red[i];
  float rstd = rsqrtf(tot2 / D + 1e-5f);
  if (t < D) o[row * (long)ldo + t] = d * rstd * w[t] + b[t];
}

// causal depthwise conv1d (K=4) + SiLU;  x rows (b*L+l), row stride ld
__global__ void k_dwconv_silu(const float* __restrict__ x, int ld,
                              const float* __restrict__ w, const float* __restrict__ b,
                              float* __restrict__ out, int B, int L, int D, long total)
{
  long idx = blockIdx.x * (long)blockDim.x + threadIdx.x;
  if (idx >= total) return;
  int d = (int)(idx % D); long t2 = idx / D;
  int l = (int)(t2 % L); int bb = (int)(t2 / L);
  float acc = b[d];
#pragma unroll
  for (int j = 0; j < 4; ++j) {
    int li = l - 3 + j;
    if (li >= 0) acc += x[((long)bb * L + li) * ld + d] * w[d * 4 + j];
  }
  out[idx] = siluf(acc);
}

// selective scan: one thread per (b, d); h[S] in registers, L sequential
__global__ void k_scan(const float* __restrict__ xc, const float* __restrict__ dt,
                       const float* __restrict__ BC, int ldbc, int offB, int offC,
                       const float* __restrict__ A, const float* __restrict__ Dp,
                       float* __restrict__ y, int B, int L, int Di, int S)
{
  int idx = blockIdx.x * blockDim.x + threadIdx.x;
  if (idx >= B * Di) return;
  int d = idx % Di, b = idx / Di;
  float h[16], Av[16];
  for (int s = 0; s < 16; ++s) h[s] = 0.f;
  for (int s = 0; s < S; ++s) Av[s] = A[d * S + s];
  float Dd = Dp[d];
  for (int l = 0; l < L; ++l) {
    long rr = (long)b * L + l;
    float xv = xc[rr * Di + d];
    float dtv = dt[rr * Di + d];
    const float* bc = BC + rr * ldbc;
    float acc = 0.f;
    for (int s = 0; s < S; ++s) {
      float a = __expf(dtv * Av[s]);
      h[s] = a * h[s] + dtv * bc[offB + s] * xv;
      acc += h[s] * bc[offC + s];
    }
    y[rr * Di + d] = acc + xv * Dd;
  }
}

__global__ void k_reverse(const float* __restrict__ in, int ldin,
                          float* __restrict__ out, int B, int L, int D, long total)
{
  long idx = blockIdx.x * (long)blockDim.x + threadIdx.x;
  if (idx >= total) return;
  int d = (int)(idx % D); long t2 = idx / D;
  int l = (int)(t2 % L); int b = (int)(t2 / L);
  out[idx] = in[((long)b * L + (L - 1 - l)) * ldin + d];
}

// out = (yf [+ yb]) * scale * silu(z)   (in-place with out==yf is safe)
__global__ void k_combine(const float* __restrict__ yf, const float* __restrict__ yb,
                          const float* __restrict__ z, int ldz, float scale,
                          float* __restrict__ out, int D, long total)
{
  long idx = blockIdx.x * (long)blockDim.x + threadIdx.x;
  if (idx >= total) return;
  int d = (int)(idx % D); long row = idx / D;
  float v = yf[idx];
  if (yb != nullptr) v += yb[idx];
  v *= scale;
  out[idx] = v * siluf(z[row * (long)ldz + d]);
}

__global__ void k_mul(float* o, const float* a, const float* b, long n) {
  long i = blockIdx.x * (long)blockDim.x + threadIdx.x;
  if (i < n) o[i] = a[i] * b[i];
}
__global__ void k_add(float* o, const float* a, const float* b, long n) {
  long i = blockIdx.x * (long)blockDim.x + threadIdx.x;
  if (i < n) o[i] = a[i] + b[i];
}
__global__ void k_add4(float* o, const float* x, const float* a,
                       const float* b, const float* c, long n) {
  long i = blockIdx.x * (long)blockDim.x + threadIdx.x;
  if (i < n) o[i] = x[i] + a[i] + b[i] + c[i];
}
__global__ void k_addscale(float* o, const float* a, const float* b, float s, long n) {
  long i = blockIdx.x * (long)blockDim.x + threadIdx.x;
  if (i < n) o[i] = (a[i] + b[i]) * s;
}
__global__ void k_sbleaky(float* x, const float* g, const float* b, int C, long n) {
  long i = blockIdx.x * (long)blockDim.x + threadIdx.x;
  if (i >= n) return;
  int c = (int)(i % C);
  x[i] = leakyf(x[i] * (g[c] * BN_S) + b[c]);
}
__global__ void k_negexp(float* o, const float* in, long n) {
  long i = blockIdx.x * (long)blockDim.x + threadIdx.x;
  if (i < n) o[i] = -__expf(in[i]);
}
__global__ void k_chexchg(const float* a, const float* b, float* oa, float* ob,
                          int D, long n) {
  long i = blockIdx.x * (long)blockDim.x + threadIdx.x;
  if (i >= n) return;
  bool m = ((i % D) & 1) == 0;
  float av = a[i], bv = b[i];
  oa[i] = m ? bv : av;
  ob[i] = m ? av : bv;
}
__global__ void k_nhwc2nchw(const float* in, float* out, int B, int H, int W, int C,
                            long n) {
  long i = blockIdx.x * (long)blockDim.x + threadIdx.x;
  if (i >= n) return;
  int w = (int)(i % W); long t = i / W;
  int h = (int)(t % H); t /= H;
  int c = (int)(t % C); int b = (int)(t / C);
  out[i] = in[(((long)b * H + h) * W + w) * C + c];
}

// ---------------------------------------------------------------------------
// Host side
// ---------------------------------------------------------------------------
#define F(i) ((const float*)d_in[(i)])
#define GRD(n) ((int)(((n) + 255) / 256))

struct Arena { char* base; size_t off; size_t cap; };
template <typename T>
static T* aalloc(Arena& a, size_t n) {
  size_t bytes = (n * sizeof(T) + 255) & ~(size_t)255;
  T* p = (T*)(a.base + a.off);
  a.off += bytes;
  return p;
}

static void gemm_h(hipStream_t st, const u16* A, int lda, const u16* W, int ldw,
                   float* C, int ldc, const float* bias, int nbias,
                   int M, int Npad, int Kpad, int act) {
  int tiles = (M >> 5) * (Npad >> 5);
  k_gemm_bf16<<<(tiles + 7) >> 3, 256, 0, st>>>(A, lda, W, ldw, C, ldc,
                                                bias, nbias, M, Npad, Kpad, act);
}

static u16* wprep(hipStream_t st, Arena& a, const float* w,
                  int N, int K, int Npad, int Kpad) {
  u16* d = aalloc<u16>(a, (size_t)Npad * Kpad);
  long tot = (long)Npad * Kpad;
  k_wcast<<<GRD(tot), 256, 0, st>>>(w, d, N, K, Npad, Kpad, tot);
  return d;
}

static u16* castbf(hipStream_t st, Arena& a, const float* src, int ld, int off,
                   long M, int K, int Kpad) {
  u16* d = aalloc<u16>(a, (size_t)M * Kpad);
  long tot = M * (long)Kpad;
  k_cast<<<GRD(tot), 256, 0, st>>>(src, ld, off, d, K, Kpad, tot);
  return d;
}

// conv: NHWC f32 in -> NHWC f32 out (bf16 WMMA implicit GEMM via im2col)
static void conv_h(hipStream_t st, Arena a, const float* in, float* out,
                   int B, int H, int W, int C, int k, int stride, int pad, int dil,
                   const u16* wbf, int Cout, int Kpad, const float* bias, int act,
                   const float* bng, const float* bnb) {
  int eff = dil * (k - 1) + 1;
  int OH = (H + 2 * pad - eff) / stride + 1;
  int OW = (W + 2 * pad - eff) / stride + 1;
  long M = (long)B * OH * OW;
  u16* col = aalloc<u16>(a, M * (size_t)Kpad);
  long tot = M * (long)Kpad;
  k_im2col<<<GRD(tot), 256, 0, st>>>(in, col, bng, bnb, B, H, W, C, OH, OW,
                                     k, k, stride, pad, dil, Kpad, tot);
  gemm_h(st, col, Kpad, wbf, Kpad, out, Cout, bias, Cout, (int)M, Cout, Kpad, act);
}

static void dgc_h(hipStream_t st, Arena a, void* const* d_in, int pg,
                  const u16* p1w, const u16* p2w, const u16* exw, const u16* pyw,
                  const float* in, float* out, int B, int C, int H, int W, int dil) {
  long M = (long)B * H * W; long n = M * C;
  int K3 = C * 9, K1 = C;
  float* t1 = aalloc<float>(a, n);
  float* t2 = aalloc<float>(a, n);
  float* c1 = aalloc<float>(a, n);
  float* c2 = aalloc<float>(a, n);
  float* c3 = aalloc<float>(a, n);
  conv_h(st, a, in, t1, B, H, W, C, 3, 1, 1, 1, p1w, C, K3, F(pg + 9), 1, F(pg + 0), F(pg + 1));
  conv_h(st, a, in, t2, B, H, W, C, 1, 1, 0, 1, p2w, C, K1, F(pg + 11), 1, F(pg + 2), F(pg + 3));
  k_mul<<<GRD(n), 256, 0, st>>>(t1, t1, t2, n);                 // xw
  conv_h(st, a, t1, t2, B, H, W, C, 3, 1, dil, dil, exw, C, K3, F(pg + 13), 1, F(pg + 4), F(pg + 5));
  k_sbleaky<<<GRD(n), 256, 0, st>>>(t2, F(pg + 6), F(pg + 7), C, n);
  conv_h(st, a, t2, c1, B, H, W, C, 3, 1, 1, 1, pyw, C, K3, F(pg + 15), 0, nullptr, nullptr);
  conv_h(st, a, c1, c2, B, H, W, C, 3, 1, 1, 1, pyw, C, K3, F(pg + 15), 0, nullptr, nullptr);
  conv_h(st, a, c2, c3, B, H, W, C, 3, 1, 1, 1, pyw, C, K3, F(pg + 15), 0, nullptr, nullptr);
  k_add4<<<GRD(n), 256, 0, st>>>(out, in, c1, c2, c3, n);
}

struct BranchW {
  const float *cw, *cb, *bdt, *Dp;
  u16 *Wx, *Wdt; float* Aneg; int S, NpadX;
};
struct MambaW {
  u16 *Win, *Wout, *Winz;
  BranchW f, b;
  bool bidir, cross, norm;
  const float *lnw, *lnb;
};

static BranchW prep_branch(hipStream_t st, Arena& per, void* const* d_in, int i0, int S) {
  BranchW bw;
  bw.cw = F(i0 + 0); bw.cb = F(i0 + 1);
  int NX = 8 + 2 * S;
  bw.NpadX = (NX + 31) & ~31;              // 24 -> 32, 40 -> 64 (32x32 GEMM tiles)
  bw.Wx = wprep(st, per, F(i0 + 2), NX, 256, bw.NpadX, 256);
  bw.Wdt = wprep(st, per, F(i0 + 3), 256, 8, 256, 32);
  bw.bdt = F(i0 + 4);
  bw.Aneg = aalloc<float>(per, (size_t)256 * S);
  long tot = 256L * S;
  k_negexp<<<GRD(tot), 256, 0, st>>>(bw.Aneg, F(i0 + 5), tot);
  bw.Dp = F(i0 + 6);
  bw.S = S;
  return bw;
}

static MambaW prep_mamba(hipStream_t st, Arena& per, void* const* d_in, int b0,
                         int S, bool bidir, bool cross, bool norm) {
  MambaW m{};
  m.Win  = wprep(st, per, F(b0 + 0), 512, 128, 512, 128);
  m.Wout = wprep(st, per, F(b0 + 1), 128, 256, 128, 256);
  m.f = prep_branch(st, per, d_in, b0 + 2, S);
  int idx = b0 + 9;
  m.bidir = bidir; m.cross = cross; m.norm = norm;
  if (bidir) { m.b = prep_branch(st, per, d_in, idx, S); idx += 7; }
  if (cross) { m.Winz = wprep(st, per, F(idx), 256, 128, 256, 128); idx += 1; }
  if (norm)  { m.lnw = F(idx); m.lnb = F(idx + 1); }
  return m;
}

static void ssm_h(hipStream_t st, Arena a, const BranchW& bw,
                  const float* xc, float* y, int B, int L) {
  long BL = (long)B * L;
  u16* xcb = castbf(st, a, xc, 256, 0, BL, 256, 256);
  float* xdbl = aalloc<float>(a, BL * (size_t)bw.NpadX);
  gemm_h(st, xcb, 256, bw.Wx, 256, xdbl, bw.NpadX, nullptr, 0, (int)BL, bw.NpadX, 256, 0);
  u16* dtb = castbf(st, a, xdbl, bw.NpadX, 0, BL, 8, 32);
  float* dt = aalloc<float>(a, BL * 256);
  gemm_h(st, dtb, 32, bw.Wdt, 32, dt, 256, bw.bdt, 256, (int)BL, 256, 32, 2);
  int nthr = B * 256;
  k_scan<<<GRD(nthr), 256, 0, st>>>(xc, dt, xdbl, bw.NpadX, 8, 8 + bw.S,
                                    bw.Aneg, bw.Dp, y, B, L, 256, bw.S);
}

static void mamba_h(hipStream_t st, Arena a, const MambaW& m,
                    const float* u, const float* extra, float* out, int B, int L) {
  long BL = (long)B * L; long n256 = BL * 256;
  u16* ub = castbf(st, a, u, 128, 0, BL, 128, 128);
  float* xz = aalloc<float>(a, BL * 512);
  gemm_h(st, ub, 128, m.Win, 128, xz, 512, nullptr, 0, (int)BL, 512, 128, 0);
  float* xc = aalloc<float>(a, n256);
  k_dwconv_silu<<<GRD(n256), 256, 0, st>>>(xz, 512, m.f.cw, m.f.cb, xc, B, L, 256, n256);
  float* yf = aalloc<float>(a, n256);
  ssm_h(st, a, m.f, xc, yf, B, L);
  float* yb = nullptr;
  if (m.bidir) {
    float* xr = aalloc<float>(a, n256);
    k_reverse<<<GRD(n256), 256, 0, st>>>(xz, 512, xr, B, L, 256, n256);
    float* xcr = aalloc<float>(a, n256);
    k_dwconv_silu<<<GRD(n256), 256, 0, st>>>(xr, 256, m.b.cw, m.b.cb, xcr, B, L, 256, n256);
    float* yrr = aalloc<float>(a, n256);
    ssm_h(st, a, m.b, xcr, yrr, B, L);
    yb = aalloc<float>(a, n256);
    k_reverse<<<GRD(n256), 256, 0, st>>>(yrr, 256, yb, B, L, 256, n256);
  }
  const float* z; int ldz;
  if (m.cross) {
    u16* eb = castbf(st, a, extra, 128, 0, BL, 128, 128);
    float* zb = aalloc<float>(a, n256);
    gemm_h(st, eb, 128, m.Winz, 128, zb, 256, nullptr, 0, (int)BL, 256, 128, 0);
    z = zb; ldz = 256;
  } else { z = xz + 256; ldz = 512; }
  float scale = m.bidir ? 0.5f : 1.0f;
  k_combine<<<GRD(n256), 256, 0, st>>>(yf, yb, z, ldz, scale, yf, 256, n256);
  if (m.norm)
    k_layernorm<<<(int)BL, 256, 0, st>>>(yf, 256, yf, 256, m.lnw, m.lnb, 256);
  u16* ybf = castbf(st, a, yf, 256, 0, BL, 256, 256);
  gemm_h(st, ybf, 256, m.Wout, 256, out, 128, nullptr, 0, (int)BL, 128, 256, 0);
}

static void single_h(hipStream_t st, Arena a, const float* nw, const float* nb,
                     const MambaW& mw, const float* x, float* out, int B, int L) {
  long BL = (long)B * L; long n = BL * 128;
  float* t = aalloc<float>(a, n);
  k_layernorm<<<(int)BL, 256, 0, st>>>(x, 128, t, 128, nw, nb, 128);
  float* mres = aalloc<float>(a, n);
  mamba_h(st, a, mw, t, nullptr, mres, B, L);
  k_add<<<GRD(n), 256, 0, st>>>(out, mres, x, n);
}

static void cross_h(hipStream_t st, Arena a, const float* n0w, const float* n0b,
                    const float* n1w, const float* n1b, const MambaW& mw,
                    const float* x0, const float* x1, float* out, int B, int L) {
  long BL = (long)B * L; long n = BL * 128;
  float* t0 = aalloc<float>(a, n);
  float* t1 = aalloc<float>(a, n);
  k_layernorm<<<(int)BL, 256, 0, st>>>(x0, 128, t0, 128, n0w, n0b, 128);
  k_layernorm<<<(int)BL, 256, 0, st>>>(x1, 128, t1, 128, n1w, n1b, 128);
  float* mres = aalloc<float>(a, n);
  mamba_h(st, a, mw, t0, t1, mres, B, L);
  k_add<<<GRD(n), 256, 0, st>>>(out, mres, x0, n);
}

struct ConvW { u16 *ci, *d2, *d3, *p1[3], *p2[3], *ex[3], *py[3]; };

static void frontend_h(hipStream_t st, Arena a, void* const* d_in, const ConvW& cw,
                       const float* img, float* seq, float* cx_out) {
  const int B = 2;
  long n0 = 2L * 256 * 256 * 32;
  float* c0 = aalloc<float>(a, n0);
  conv_h(st, a, img, c0, B, 256, 256, 1, 3, 1, 1, 1, cw.ci, 32, 32, F(3), 0, nullptr, nullptr);
  k_nhwc2nchw<<<GRD(n0), 256, 0, st>>>(c0, cx_out, B, 256, 256, 32, n0);
  float* g1 = aalloc<float>(a, n0);
  dgc_h(st, a, d_in, 4, cw.p1[0], cw.p2[0], cw.ex[0], cw.py[0], c0, g1, B, 32, 256, 256, 1);
  long n1 = 2L * 128 * 128 * 64;
  float* h2 = aalloc<float>(a, n1);
  conv_h(st, a, g1, h2, B, 256, 256, 32, 2, 2, 0, 1, cw.d2, 64, 128, F(21), 0, nullptr, nullptr);
  float* g2 = aalloc<float>(a, n1);
  dgc_h(st, a, d_in, 22, cw.p1[1], cw.p2[1], cw.ex[1], cw.py[1], h2, g2, B, 64, 128, 128, 3);
  long n2 = 2L * 64 * 64 * 128;
  float* h3 = aalloc<float>(a, n2);
  conv_h(st, a, g2, h3, B, 128, 128, 64, 2, 2, 0, 1, cw.d3, 128, 256, F(39), 0, nullptr, nullptr);
  dgc_h(st, a, d_in, 40, cw.p1[2], cw.p2[2], cw.ex[2], cw.py[2], h3, seq, B, 128, 64, 64, 5);
}

extern "C" void kernel_launch(void* const* d_in, const int* in_sizes, int n_in,
                              void* d_out, int out_size, void* d_ws, size_t ws_size,
                              hipStream_t stream) {
  (void)in_sizes; (void)n_in; (void)out_size;
  hipStream_t st = stream;
  Arena per{ (char*)d_ws, 0, ws_size };

  // ---- weight prep (bf16, padded) ----
  ConvW cw;
  cw.ci = wprep(st, per, F(2), 32, 9, 32, 32);
  cw.d2 = wprep(st, per, F(20), 64, 128, 64, 128);
  cw.d3 = wprep(st, per, F(38), 128, 256, 128, 256);
  const int dbase[3] = {4, 22, 40};
  const int dch[3]   = {32, 64, 128};
  for (int i = 0; i < 3; ++i) {
    int b = dbase[i], C = dch[i];
    cw.p1[i] = wprep(st, per, F(b + 8),  C, 9 * C, C, 9 * C);
    cw.p2[i] = wprep(st, per, F(b + 10), C, C,     C, C);
    cw.ex[i] = wprep(st, per, F(b + 12), C, 9 * C, C, 9 * C);
    cw.py[i] = wprep(st, per, F(b + 14), C, 9 * C, C, 9 * C);
  }
  MambaW m_xm0 = prep_mamba(st, per, d_in, 58, 8, true, false, true);
  MambaW m_ym0 = prep_mamba(st, per, d_in, 78, 8, true, false, true);
  MambaW m_e1  = prep_mamba(st, per, d_in, 100, 16, false, false, false);
  MambaW m_e2  = prep_mamba(st, per, d_in, 109, 16, false, false, false);
  MambaW m_xc  = prep_mamba(st, per, d_in, 122, 8, true, true, true);
  MambaW m_yc  = prep_mamba(st, per, d_in, 145, 8, true, true, true);
  u16* Wfin = wprep(st, per, F(164), 128, 128, 128, 128);

  const int B = 2, L = 4096;
  const long BL = (long)B * L, n128 = BL * 128;
  const long SEQ = n128;                 // 1048576
  const long IMG32 = 2L * 32 * 256 * 256;
  float* fo = (float*)d_out;
  float* out_fus = fo;
  float* out_xf = fo + SEQ;
  float* out_yf = fo + 2 * SEQ;
  float* out_cx = fo + 3 * SEQ;
  float* out_cy = fo + 3 * SEQ + IMG32;

  // ---- conv front-end (x then y reuse the same scratch) ----
  float* xseq = aalloc<float>(per, n128);
  float* yseq = aalloc<float>(per, n128);
  frontend_h(st, per, d_in, cw, (const float*)d_in[0], xseq, out_cx);
  frontend_h(st, per, d_in, cw, (const float*)d_in[1], yseq, out_cy);

  // ---- mamba stage ----
  float* xm = aalloc<float>(per, n128);
  float* ym = aalloc<float>(per, n128);
  single_h(st, per, F(56), F(57), m_xm0, xseq, xm, B, L);
  single_h(st, per, F(76), F(77), m_ym0, yseq, ym, B, L);

  const float* I1 = xm; const float* I2 = ym;
  const float* r1 = xseq; const float* r2 = yseq;
  for (int it = 0; it < 2; ++it) {
    float* r1n = aalloc<float>(per, n128);
    float* r2n = aalloc<float>(per, n128);
    float* o1  = aalloc<float>(per, n128);
    float* o2  = aalloc<float>(per, n128);
    Arena a = per;  // scratch above persistent watermark
    k_add<<<GRD(n128), 256, 0, st>>>(r1n, I1, r1, n128);
    k_add<<<GRD(n128), 256, 0, st>>>(r2n, I2, r2, n128);
    float* I1n = aalloc<float>(a, n128);
    float* I2n = aalloc<float>(a, n128);
    k_layernorm<<<(int)BL, 256, 0, st>>>(r1n, 128, I1n, 128, F(96), F(97), 128);
    k_layernorm<<<(int)BL, 256, 0, st>>>(r2n, 128, I2n, 128, F(98), F(99), 128);
    float* I1s = aalloc<float>(a, n128);
    float* I2s = aalloc<float>(a, n128);
    k_chexchg<<<GRD(n128), 256, 0, st>>>(I1n, I2n, I1s, I2s, 128, n128);
    mamba_h(st, a, m_e1, I1s, nullptr, o1, B, L);
    mamba_h(st, a, m_e2, I2s, nullptr, o2, B, L);
    I1 = o1; I2 = o2; r1 = r1n; r2 = r2n;
  }

  cross_h(st, per, F(118), F(119), F(120), F(121), m_xc, I1, I2, out_xf, B, L);
  cross_h(st, per, F(141), F(142), F(143), F(144), m_yc, I2, I1, out_yf, B, L);

  // ---- fusion head ----
  Arena a = per;
  float* fsum = aalloc<float>(a, n128);
  k_addscale<<<GRD(n128), 256, 0, st>>>(fsum, out_xf, out_yf, 0.5f, n128);
  u16* fb = castbf(st, a, fsum, 128, 0, BL, 128, 128);
  gemm_h(st, fb, 128, Wfin, 128, out_fus, 128, F(165), 128, (int)BL, 128, 128, 0);
}
